// MultiHeadAttention_15659450761618
// MI455X (gfx1250) — compile-verified, matrix operand order checked by reading, not run on hardware
//
#include <hip/hip_runtime.h>
#include <hip/hip_bf16.h>

typedef __attribute__((ext_vector_type(16))) __bf16 v16bf;
typedef __attribute__((ext_vector_type(8)))  __bf16 v8bf;
typedef __attribute__((ext_vector_type(8)))  float  v8f;
typedef __attribute__((ext_vector_type(4)))  int    i32x4;

using u16 = unsigned short;

// ---------------------------------------------------------------- constants
#define BATCH   2
#define SEQ     4096
#define DMODEL  512
#define NHEADS  8
#define DEPTH   64
#define BH      (BATCH * NHEADS)          // 16
#define MROWS   (BATCH * SEQ)             // 8192
#define OUT_ELEMS ((size_t)BATCH * SEQ * DMODEL)  // 4,194,304 floats

// ---------------------------------------------------------------- helpers
static __device__ __forceinline__ u16 f2bf(float f) {
  unsigned u = __builtin_bit_cast(unsigned, f);
  unsigned r = u + 0x7FFFu + ((u >> 16) & 1u);   // round-to-nearest-even
  return (u16)(r >> 16);
}
static __device__ __forceinline__ unsigned pk2bf(float a, float b) {
  return (unsigned)f2bf(a) | ((unsigned)f2bf(b) << 16);
}
static __device__ __forceinline__ v16bf pack16(v8bf lo, v8bf hi) {
  v16bf r;
#pragma unroll
  for (int i = 0; i < 8; ++i) { r[i] = lo[i]; r[i + 8] = hi[i]; }
  return r;
}
// Load a 16-elem bf16 A/B fragment slice: p points at (k + half*8); the upper
// 8 elements live at +16 elements (= k + 16 + half*8). 16B-aligned loads.
static __device__ __forceinline__ v16bf load_frag(const u16* p) {
  v8bf lo = *reinterpret_cast<const v8bf*>(p);
  v8bf hi = *reinterpret_cast<const v8bf*>(p + 16);
  return pack16(lo, hi);
}
static __device__ __forceinline__ v8f wmma_bf16(v16bf a, v16bf b, v8f c) {
  return __builtin_amdgcn_wmma_f32_16x16x32_bf16(false, a, false, b,
                                                 (short)0, c, false, false);
}
static __device__ __forceinline__ v8f zero8() {
  v8f z = {0.f, 0.f, 0.f, 0.f, 0.f, 0.f, 0.f, 0.f};
  return z;
}
// 16-wide (half-wave) butterfly reductions; rows of a 16x16 WMMA C tile live
// in 16-lane groups, so xor masks 1/2/4/8 stay inside the group.
static __device__ __forceinline__ float redmax16(float v) {
  v = fmaxf(v, __shfl_xor(v, 1, 32));
  v = fmaxf(v, __shfl_xor(v, 2, 32));
  v = fmaxf(v, __shfl_xor(v, 4, 32));
  v = fmaxf(v, __shfl_xor(v, 8, 32));
  return v;
}
static __device__ __forceinline__ float redsum16(float v) {
  v += __shfl_xor(v, 1, 32);
  v += __shfl_xor(v, 2, 32);
  v += __shfl_xor(v, 4, 32);
  v += __shfl_xor(v, 8, 32);
  return v;
}

// 16-byte global -> LDS copy. Uses the CDNA5 async-to-LDS path when the
// builtin exists (tracked by ASYNCcnt), else a plain load+ds_store fallback.
#define HAVE_ASYNC_LDS __has_builtin(__builtin_amdgcn_global_load_async_to_lds_b128)
static __device__ __forceinline__ void copy16_to_lds(u16* dst_lds, const u16* src) {
#if HAVE_ASYNC_LDS
  // Param types per hipcc diagnostic: (i32x4 as1*, i32x4 as3*, imm, imm).
  // Flat->as1: identity. Flat->as3: low 32 bits are the LDS byte offset
  typedef __attribute__((address_space(1))) i32x4* as1p;
  typedef __attribute__((address_space(3))) i32x4* as3p;
  __builtin_amdgcn_global_load_async_to_lds_b128(
      (as1p)(unsigned long long)src,
      (as3p)(unsigned)(unsigned long long)dst_lds, 0, 0);
#else
  *reinterpret_cast<v8bf*>(dst_lds) = *reinterpret_cast<const v8bf*>(src);
#endif
}
static __device__ __forceinline__ void copy_wait() {
#if HAVE_ASYNC_LDS
  asm volatile("s_wait_asynccnt 0x0" ::: "memory");
#endif
}

// ---------------------------------------------------------------- kernel 1
// Convert q,k,v activations f32 -> bf16 (vectorized: float4 in, 8B out).
__global__ void __launch_bounds__(256)
mha_act_kernel(const float* __restrict__ q, const float* __restrict__ k,
               const float* __restrict__ v, u16* __restrict__ qb,
               u16* __restrict__ kb, u16* __restrict__ vb) {
  int gid = blockIdx.x * 256 + threadIdx.x;        // 1,048,576 float4 groups
  float4 fq = reinterpret_cast<const float4*>(q)[gid];
  float4 fk = reinterpret_cast<const float4*>(k)[gid];
  float4 fv = reinterpret_cast<const float4*>(v)[gid];
  uint2 rq = {pk2bf(fq.x, fq.y), pk2bf(fq.z, fq.w)};
  uint2 rk = {pk2bf(fk.x, fk.y), pk2bf(fk.z, fk.w)};
  uint2 rv = {pk2bf(fv.x, fv.y), pk2bf(fv.z, fv.w)};
  reinterpret_cast<uint2*>(qb)[gid] = rq;
  reinterpret_cast<uint2*>(kb)[gid] = rk;
  reinterpret_cast<uint2*>(vb)[gid] = rv;
}

// ---------------------------------------------------------------- kernel 2
// Convert + transpose the four 512x512 f32 weight matrices into bf16 Wt[N][K].
__global__ void __launch_bounds__(256)
mha_wt_kernel(const float* __restrict__ w0, const float* __restrict__ w1,
              const float* __restrict__ w2, const float* __restrict__ w3,
              u16* __restrict__ wt) {
  int gid = blockIdx.x * 256 + threadIdx.x;        // 4*512*128 threads
  int mat = gid >> 16;
  int rem = gid & 65535;
  int n = rem >> 7;
  int kq = (rem & 127) * 4;
  const float* w = (mat == 0) ? w0 : (mat == 1) ? w1 : (mat == 2) ? w2 : w3;
  float a0 = w[(kq + 0) * 512 + n];
  float a1 = w[(kq + 1) * 512 + n];
  float a2 = w[(kq + 2) * 512 + n];
  float a3 = w[(kq + 3) * 512 + n];
  uint2 r = {pk2bf(a0, a1), pk2bf(a2, a3)};
  reinterpret_cast<uint2*>(wt)[((size_t)mat * 262144 + n * 512 + kq) >> 2] = r;
}

// ---------------------------------------------------------------- kernel 3
// C[M=8192, N=512] = Xbf @ W + b.  One 16x16 tile per wave, K=512 in 32-steps.
// MODE 0: store bf16 head-split  dst[((b*H+h)*S + s)*64 + d]   (Q, K)
// MODE 1: store bf16 transposed  dst[((b*H+h)*64 + d)*S + s]   (V)
template <int MODE>
__global__ void __launch_bounds__(128)
mha_proj_kernel(const u16* __restrict__ X, const u16* __restrict__ Wt,
                const float* __restrict__ bias, u16* __restrict__ dst) {
  const int wave = threadIdx.x >> 5;
  const int lane = threadIdx.x & 31;
  const int ln = lane & 15;
  const int hf = lane >> 4;
  const int mtile = blockIdx.x >> 3;               // 512 tiles of 16 rows
  const int ntile = (blockIdx.x & 7) * 4 + wave;   // 32 tiles of 16 cols
  const int m0 = mtile * 16, n0 = ntile * 16;

  const u16* xrow = X + (size_t)(m0 + ln) * DMODEL;
  const u16* wrow = Wt + (size_t)(n0 + ln) * DMODEL;

  v8f acc = zero8();
#pragma unroll 4
  for (int k = 0; k < DMODEL; k += 32) {
    v16bf A = load_frag(xrow + k + hf * 8);
    v16bf B = load_frag(wrow + k + hf * 8);
    acc = wmma_bf16(A, B, acc);
  }

  const float bv = bias[n0 + ln];
  const int col = n0 + ln;
  const int h = col >> 6, d = col & 63;
#pragma unroll
  for (int r = 0; r < 8; ++r) {
    int row = m0 + hf * 8 + r;                     // global row in [0,8192)
    int b = row >> 12, s = row & 4095;
    u16 bf = f2bf(acc[r] + bv);
    if constexpr (MODE == 0)
      dst[(((size_t)(b * NHEADS + h) * SEQ) + s) * DEPTH + d] = bf;
    else
      dst[(((size_t)(b * NHEADS + h) * DEPTH) + d) * SEQ + s] = bf;
  }
}

// ---------------------------------------------------------------- kernel 4
// Attention: 4 waves/block, all on the same (b,h); wave w owns q-tile
// blockIdx*4+w. K/V chunks staged once per block in LDS (async-to-LDS),
// two-pass online softmax in the exp2 domain, full f32 attn written,
// ctx accumulated via WMMA.
__global__ void __launch_bounds__(128)
mha_attn_kernel(const u16* __restrict__ qh, const u16* __restrict__ kh,
                const u16* __restrict__ vt, u16* __restrict__ ctx,
                float* __restrict__ attn) {
  __shared__ alignas(16) u16 ldsK[32 * 64];        // 32 keys x 64 depth (4KB)
  __shared__ alignas(16) u16 ldsV[64 * 32];        // 64 depth x 32 keys (4KB)
  __shared__ alignas(16) u16 ldsP[4][16 * 32];     // per-wave P staging (4KB)

  const int tid  = threadIdx.x;
  const int wave = tid >> 5;
  const int lane = tid & 31;
  const int ln = lane & 15;
  const int hf = lane >> 4;

  const int tile = blockIdx.x * 4 + wave;          // 0..4095
  const int qt = tile & 255;
  const int bh = tile >> 8;                        // same for all 4 waves
  const int q0 = qt * 16;

  // exp2-domain scale: logit * (1/sqrt(64)) * log2(e)
  const float c = 0.125f * 1.44269504088896f;

  // Q fragments: rows q0..q0+15, K = depth 64 (two 32-chunks)
  const u16* qrow = qh + ((size_t)bh * SEQ + q0 + ln) * DEPTH;
  v16bf aq0 = load_frag(qrow + 0  + hf * 8);
  v16bf aq1 = load_frag(qrow + 32 + hf * 8);

  const u16* kbase_p = kh + (size_t)bh * SEQ * DEPTH;
  const u16* vbase_p = vt + (size_t)bh * DEPTH * SEQ;

  float Mr[8], Sr[8];
#pragma unroll
  for (int r = 0; r < 8; ++r) { Mr[r] = -1e30f; Sr[r] = 0.f; }

  // ---- pass 1: running row max / sum (exp2 domain) ------------------
  for (int kbase = 0; kbase < SEQ; kbase += 32) {
    // stage K chunk [32 keys][64 depth] = 512 x 16B, 4 per thread
    const u16* ksrc = kbase_p + (size_t)kbase * DEPTH;
#pragma unroll
    for (int j = 0; j < 4; ++j) {
      int cc = j * 128 + tid;
      int row = cc >> 3, seg = cc & 7;
      copy16_to_lds(&ldsK[row * 64 + seg * 8], ksrc + row * 64 + seg * 8);
    }
    if (kbase + 32 < SEQ)   // prefetch next K chunk (global_prefetch_b8)
      __builtin_prefetch(ksrc + 32 * DEPTH + tid * 16, 0, 1);
    copy_wait();
    __syncthreads();

    // both 16-key tiles of the chunk first, then ONE reduction per row
    const u16* kr0 = &ldsK[(0 * 16 + ln) * 64];
    const u16* kr1 = &ldsK[(1 * 16 + ln) * 64];
    v8f acc0 = zero8(), acc1 = zero8();
    acc0 = wmma_bf16(aq0, load_frag(kr0 + 0  + hf * 8), acc0);
    acc0 = wmma_bf16(aq1, load_frag(kr0 + 32 + hf * 8), acc0);
    acc1 = wmma_bf16(aq0, load_frag(kr1 + 0  + hf * 8), acc1);
    acc1 = wmma_bf16(aq1, load_frag(kr1 + 32 + hf * 8), acc1);
#pragma unroll
    for (int r = 0; r < 8; ++r) {
      float v0 = acc0[r] * c;
      float v1 = acc1[r] * c;
      float tmax = redmax16(fmaxf(v0, v1));        // 32-key row max
      float newM = fmaxf(Mr[r], tmax);
      float ps = redsum16(exp2f(v0 - newM) + exp2f(v1 - newM));
      Sr[r] = Sr[r] * exp2f(Mr[r] - newM) + ps;
      Mr[r] = newM;
    }
    __syncthreads();
  }
  float inv[8];
#pragma unroll
  for (int r = 0; r < 8; ++r) inv[r] = 1.0f / Sr[r];

  // ---- pass 2: write attn, accumulate ctx ---------------------------
  v8f cacc0 = zero8(), cacc1 = zero8(), cacc2 = zero8(), cacc3 = zero8();
  float* attn_base = attn + ((size_t)bh * SEQ + q0) * SEQ;

  for (int kbase = 0; kbase < SEQ; kbase += 32) {
    // stage K chunk + V chunk
    const u16* ksrc = kbase_p + (size_t)kbase * DEPTH;
#pragma unroll
    for (int j = 0; j < 4; ++j) {
      int cc = j * 128 + tid;
      int row = cc >> 3, seg = cc & 7;
      copy16_to_lds(&ldsK[row * 64 + seg * 8], ksrc + row * 64 + seg * 8);
    }
    const u16* vsrc = vbase_p + kbase;
#pragma unroll
    for (int j = 0; j < 2; ++j) {
      int cc = j * 128 + tid;
      int row = cc >> 2, seg = cc & 3;
      copy16_to_lds(&ldsV[row * 32 + seg * 8], vsrc + (size_t)row * SEQ + seg * 8);
    }
    if (kbase + 32 < SEQ) {
      __builtin_prefetch(ksrc + 32 * DEPTH + tid * 16, 0, 1);
      if (tid < 64) __builtin_prefetch(vsrc + (size_t)tid * SEQ + 32, 0, 1);
    }
    copy_wait();
    __syncthreads();

#pragma unroll
    for (int t = 0; t < 2; ++t) {
      const int kb = kbase + t * 16;
      const u16* krow = &ldsK[(t * 16 + ln) * 64];
      v8f acc = zero8();
      acc = wmma_bf16(aq0, load_frag(krow + 0  + hf * 8), acc);
      acc = wmma_bf16(aq1, load_frag(krow + 32 + hf * 8), acc);
#pragma unroll
      for (int r = 0; r < 8; ++r) {
        float p = exp2f(acc[r] * c - Mr[r]) * inv[r];
        int row = hf * 8 + r;
        attn_base[(size_t)row * SEQ + kb + ln] = p;
        ldsP[wave][row * 32 + t * 16 + ln] = f2bf(p);  // same-wave, in-order
      }
    }
    // P as WMMA A fragment (16 rows x 32 keys) from this wave's LDS region
    const u16* lrow = &ldsP[wave][ln * 32];
    v16bf Ap = load_frag(lrow + hf * 8);
    // ctx += P x V  (B fragments from staged transposed V, contiguous keys)
    cacc0 = wmma_bf16(Ap, load_frag(&ldsV[(0 * 16 + ln) * 32 + hf * 8]), cacc0);
    cacc1 = wmma_bf16(Ap, load_frag(&ldsV[(1 * 16 + ln) * 32 + hf * 8]), cacc1);
    cacc2 = wmma_bf16(Ap, load_frag(&ldsV[(2 * 16 + ln) * 32 + hf * 8]), cacc2);
    cacc3 = wmma_bf16(Ap, load_frag(&ldsV[(3 * 16 + ln) * 32 + hf * 8]), cacc3);
    __syncthreads();
  }

  // ---- store bf16 ctx in concat layout [B*S, 512] -------------------
  const int b = bh >> 3, h = bh & 7;
#pragma unroll
  for (int r = 0; r < 8; ++r) {
    const size_t rowoff = ((size_t)(b * SEQ + q0 + hf * 8 + r)) * DMODEL + h * DEPTH;
    ctx[rowoff + 0 * 16 + ln] = f2bf(cacc0[r]);
    ctx[rowoff + 1 * 16 + ln] = f2bf(cacc1[r]);
    ctx[rowoff + 2 * 16 + ln] = f2bf(cacc2[r]);
    ctx[rowoff + 3 * 16 + ln] = f2bf(cacc3[r]);
  }
}

// ---------------------------------------------------------------- kernel 5
// out[M=8192,512] = ctx_bf16 @ wo + b  (f32 output)
__global__ void __launch_bounds__(128)
mha_outproj_kernel(const u16* __restrict__ ctx, const u16* __restrict__ Wt,
                   const float* __restrict__ bias, float* __restrict__ out) {
  const int wave = threadIdx.x >> 5;
  const int lane = threadIdx.x & 31;
  const int ln = lane & 15;
  const int hf = lane >> 4;
  const int mtile = blockIdx.x >> 3;
  const int ntile = (blockIdx.x & 7) * 4 + wave;
  const int m0 = mtile * 16, n0 = ntile * 16;

  const u16* arow = ctx + (size_t)(m0 + ln) * DMODEL;
  const u16* wrow = Wt + (size_t)(n0 + ln) * DMODEL;

  v8f acc = zero8();
#pragma unroll 4
  for (int k = 0; k < DMODEL; k += 32) {
    v16bf A = load_frag(arow + k + hf * 8);
    v16bf B = load_frag(wrow + k + hf * 8);
    acc = wmma_bf16(A, B, acc);
  }
  const float bv = bias[n0 + ln];
#pragma unroll
  for (int r = 0; r < 8; ++r) {
    int row = m0 + hf * 8 + r;
    out[(size_t)row * DMODEL + n0 + ln] = acc[r] + bv;
  }
}

// ---------------------------------------------------------------- launch
extern "C" void kernel_launch(void* const* d_in, const int* in_sizes, int n_in,
                              void* d_out, int out_size, void* d_ws, size_t ws_size,
                              hipStream_t stream) {
  (void)in_sizes; (void)n_in; (void)out_size; (void)ws_size;

  const float* v_in  = (const float*)d_in[0];
  const float* k_in  = (const float*)d_in[1];
  const float* q_in  = (const float*)d_in[2];
  const float* wq_w  = (const float*)d_in[3];
  const float* wq_b  = (const float*)d_in[4];
  const float* wk_w  = (const float*)d_in[5];
  const float* wk_b  = (const float*)d_in[6];
  const float* wv_w  = (const float*)d_in[7];
  const float* wv_b  = (const float*)d_in[8];
  const float* wo_w  = (const float*)d_in[9];
  const float* wo_b  = (const float*)d_in[10];

  // workspace (u16 elems): Wt[4*512*512], qbf/kbf/vbf, Qh, Kh, Vt, ctx
  const size_t WT_ELEMS  = (size_t)4 * 512 * 512;              // 1,048,576
  const size_t ACT_ELEMS = (size_t)MROWS * DMODEL;             // 4,194,304
  u16* wt  = (u16*)d_ws;
  u16* qbf = wt  + WT_ELEMS;
  u16* kbf = qbf + ACT_ELEMS;
  u16* vbf = kbf + ACT_ELEMS;
  u16* qh  = vbf + ACT_ELEMS;
  u16* kh  = qh  + ACT_ELEMS;
  u16* vtp = kh  + ACT_ELEMS;
  u16* ctx = vtp + ACT_ELEMS;

  float* out_f  = (float*)d_out;
  float* attn_f = out_f + OUT_ELEMS;

  // 1) activation bf16 convert + weight transpose/convert
  mha_act_kernel<<<4096, 256, 0, stream>>>(q_in, k_in, v_in, qbf, kbf, vbf);
  mha_wt_kernel<<<1024, 256, 0, stream>>>(wq_w, wk_w, wv_w, wo_w, wt);

  // 2) QKV projections (512 mtiles * 8 ngroups = 4096 blocks, 4 waves each)
  mha_proj_kernel<0><<<4096, 128, 0, stream>>>(qbf, wt + 0 * 262144, wq_b, qh);
  mha_proj_kernel<0><<<4096, 128, 0, stream>>>(kbf, wt + 1 * 262144, wk_b, kh);
  mha_proj_kernel<1><<<4096, 128, 0, stream>>>(vbf, wt + 2 * 262144, wv_b, vtp);

  // 3) attention: 4096 (b,h,qtile) waves / 4 per block
  mha_attn_kernel<<<1024, 128, 0, stream>>>(qh, kh, vtp, ctx, attn_f);

  // 4) output projection
  mha_outproj_kernel<<<4096, 128, 0, stream>>>(ctx, wt + 3 * 262144, wo_b, out_f);
}
